// yanet_88682484728478
// MI455X (gfx1250) — matrix operand order repaired and checked by reading, not compiled, not run on hardware
//
#include <hip/hip_runtime.h>
#include <hip/hip_bf16.h>

typedef __attribute__((ext_vector_type(16))) _Float16 v16h;
typedef __attribute__((ext_vector_type(8)))  float    v8f;

#define TM 128
#define TN 64
#define TK 32
#define LSTR (TK + 8)   // 40 halves = 80B row stride -> 16B-aligned b128 LDS accesses

// ---------------------------------------------------------------------------
// Implicit-GEMM 3x3 SAME conv on NHWC f16 activations (C multiple of 32).
// Tap-major K: every TK=32 chunk sits inside one filter tap, so halo bounds
// are uniform per thread per tap and the channel slab is contiguous.
// ALL LDS staging (A and B) uses global_load_async_to_lds_b128 (ASYNCcnt),
// double-buffered and software-pipelined: DMA of chunk q+1 overlaps the
// fragment reads + 4x v_wmma_f32_16x16x32_f16 of chunk q.
// ---------------------------------------------------------------------------
__global__ __launch_bounds__(256) void conv3x3_wmma(
    const _Float16* __restrict__ in, const _Float16* __restrict__ wkT,
    const float* __restrict__ bias, const _Float16* __restrict__ resid,
    _Float16* __restrict__ out,
    int N, int Cin, int H, int W, int Cout, int relu)
{
  const int HW   = H * W;
  const int Npix = N * HW;
  const int tileM = blockIdx.x * TM;
  const int tileN = blockIdx.y * TN;
  const int t    = threadIdx.x;
  const int lane = t & 31;
  const int wave = t >> 5;

  __shared__ __align__(16) _Float16 As[2][TM][LSTR];
  __shared__ __align__(16) _Float16 Bs[2][TN][LSTR];   // Bs[b][col][k]
  __shared__ __align__(16) _Float16 dump[16];          // halo async-DMA landfill

  // A loader: 2 threads per pixel; each owns a contiguous 16-channel (32B) slab.
  const int am   = t & 127;
  const int koff = (t >> 7) * 16;
  int an = 0, ay = 0, ax = 0;
  const int gp = tileM + am;
  const bool avalid = gp < Npix;
  if (avalid) { an = gp / HW; int rem = gp - an * HW; ay = rem / W; ax = rem - ay * W; }

  // B loader: 4 threads per col; one 16B async load each.
  const int bcol  = t >> 2;
  const int bcoff = (t & 3) * 8;
  const uint32_t blds0 = (uint32_t)(uintptr_t)&Bs[0][bcol][bcoff];
  const uint32_t blds1 = (uint32_t)(uintptr_t)&Bs[1][bcol][bcoff];
  const uint32_t dumpa = (uint32_t)(uintptr_t)&dump[0];

  // wave -> M-blocks {2a,2a+1}, N-blocks {2b,2b+1}
  const int mb0 = (wave & 3) * 2;
  const int nb0 = (wave >> 2) * 2;

  v8f acc[2][2];
#pragma unroll
  for (int i = 0; i < 2; ++i)
#pragma unroll
    for (int j = 0; j < 2; ++j)
      acc[i][j] = (v8f){0.f, 0.f, 0.f, 0.f, 0.f, 0.f, 0.f, 0.f};

  const int nq = Cin / TK;

  for (int ky = 0; ky < 3; ++ky) {
    const int iy = ay + ky - 1;
    for (int kx = 0; kx < 3; ++kx) {
      const int ix  = ax + kx - 1;
      const bool pok = avalid && (unsigned)iy < (unsigned)H && (unsigned)ix < (unsigned)W;
      const _Float16* asrc = in + ((size_t)((size_t)an * H + iy) * W + ix) * Cin + koff;
      const int rtap = ky * 3 + kx;
      const _Float16* bsrc = wkT + ((size_t)rtap * Cout + tileN + bcol) * Cin + bcoff;

      __syncthreads();   // previous tap's fragment reads are done before restaging

      // effective A DMA endpoints: halo lanes dump to scratch LDS from a safe address
      const _Float16* aeff = pok ? asrc : (in + koff);
      const uint32_t la0 = pok ? (uint32_t)(uintptr_t)&As[0][am][koff] : dumpa;
      const uint32_t la1 = pok ? (uint32_t)(uintptr_t)&As[1][am][koff] : dumpa;
      if (!pok) {   // this thread's real slabs stay zero for the whole tap
        const uint4 z = {0u, 0u, 0u, 0u};
        uint4* d0 = (uint4*)&As[0][am][koff];
        uint4* d1 = (uint4*)&As[1][am][koff];
        d0[0] = z; d0[1] = z;
        d1[0] = z; d1[1] = z;
      }

      // prologue: DMA chunk 0 into buffer 0 (A: 32B via paired offsets, B: 16B)
      {
        const uint64_t ga = (uint64_t)(uintptr_t)aeff;
        asm volatile("global_load_async_to_lds_b128 %0, %1, off\n\t"
                     "global_load_async_to_lds_b128 %0, %1, off offset:16"
                     :: "v"(la0), "v"(ga) : "memory");
        const uint64_t gb = (uint64_t)(uintptr_t)bsrc;
        asm volatile("global_load_async_to_lds_b128 %0, %1, off"
                     :: "v"(blds0), "v"(gb) : "memory");
      }

      for (int q = 0; q < nq; ++q) {
        asm volatile("s_wait_asynccnt 0" ::: "memory");   // chunk q landed in LDS
        __syncthreads();

        if (q + 1 < nq) {   // DMA chunk q+1 while computing chunk q
          const int c1 = (q + 1) * TK;
          const uint64_t ga = (uint64_t)(uintptr_t)(aeff + c1);
          const uint32_t la = (q & 1) ? la0 : la1;
          asm volatile("global_load_async_to_lds_b128 %0, %1, off\n\t"
                       "global_load_async_to_lds_b128 %0, %1, off offset:16"
                       :: "v"(la), "v"(ga) : "memory");
          const uint64_t gb = (uint64_t)(uintptr_t)(bsrc + c1);
          const uint32_t lb = (q & 1) ? blds0 : blds1;
          asm volatile("global_load_async_to_lds_b128 %0, %1, off"
                       :: "v"(lb), "v"(gb) : "memory");
        }

        const int qb = q & 1;
        // fragments (ISA layouts) + 4 WMMAs
        v16h af[2];
#pragma unroll
        for (int i = 0; i < 2; ++i) {
          const int row = (mb0 + i) * 16 + (lane & 15);
          const int kb  = (lane >> 4) ? 8 : 0;
#pragma unroll
          for (int h = 0; h < 8; ++h) af[i][h] = As[qb][row][kb + h];
#pragma unroll
          for (int h = 0; h < 8; ++h) af[i][8 + h] = As[qb][row][kb + 16 + h];
        }
        v16h bf[2];
#pragma unroll
        for (int j = 0; j < 2; ++j) {
          const int col = (nb0 + j) * 16 + (lane & 15);
          const int k0  = (lane >> 4) ? 8 : 0;
#pragma unroll
          for (int h = 0; h < 8; ++h) bf[j][h] = Bs[qb][col][k0 + h];
#pragma unroll
          for (int h = 0; h < 8; ++h) bf[j][8 + h] = Bs[qb][col][k0 + 16 + h];
        }
        acc[0][0] = __builtin_amdgcn_wmma_f32_16x16x32_f16(false, af[0], false, bf[0], (short)0, acc[0][0], false, false);
        acc[0][1] = __builtin_amdgcn_wmma_f32_16x16x32_f16(false, af[0], false, bf[1], (short)0, acc[0][1], false, false);
        acc[1][0] = __builtin_amdgcn_wmma_f32_16x16x32_f16(false, af[1], false, bf[0], (short)0, acc[1][0], false, false);
        acc[1][1] = __builtin_amdgcn_wmma_f32_16x16x32_f16(false, af[1], false, bf[1], (short)0, acc[1][1], false, false);
      }
    }
  }

  // epilogue (NHWC): VGPR r -> M=r (lanes 0-15) / M=8+r (lanes 16-31); N = lane&15
  const float lo = relu ? 0.0f : -__builtin_inff();   // branchless uniform ReLU
  if (resid) {
#pragma unroll
    for (int i = 0; i < 2; ++i)
#pragma unroll
      for (int j = 0; j < 2; ++j) {
        const int co = tileN + (nb0 + j) * 16 + (lane & 15);
        const float bv = bias[co];
        const int rowAdd = (lane >> 4) * 8;
#pragma unroll
        for (int r = 0; r < 8; ++r) {
          const int gpix = tileM + (mb0 + i) * 16 + rowAdd + r;
          if (gpix < Npix) {
            const size_t oidx = (size_t)gpix * Cout + co;
            out[oidx] = (_Float16)fmaxf(acc[i][j][r] + bv + (float)resid[oidx], lo);
          }
        }
      }
  } else if (bias) {
#pragma unroll
    for (int i = 0; i < 2; ++i)
#pragma unroll
      for (int j = 0; j < 2; ++j) {
        const int co = tileN + (nb0 + j) * 16 + (lane & 15);
        const float bv = bias[co];
        const int rowAdd = (lane >> 4) * 8;
#pragma unroll
        for (int r = 0; r < 8; ++r) {
          const int gpix = tileM + (mb0 + i) * 16 + rowAdd + r;
          if (gpix < Npix)
            out[(size_t)gpix * Cout + co] = (_Float16)fmaxf(acc[i][j][r] + bv, lo);
        }
      }
  } else {
#pragma unroll
    for (int i = 0; i < 2; ++i)
#pragma unroll
      for (int j = 0; j < 2; ++j) {
        const int co = tileN + (nb0 + j) * 16 + (lane & 15);
        const int rowAdd = (lane >> 4) * 8;
#pragma unroll
        for (int r = 0; r < 8; ++r) {
          const int gpix = tileM + (mb0 + i) * 16 + rowAdd + r;
          if (gpix < Npix)
            out[(size_t)gpix * Cout + co] = (_Float16)fmaxf(acc[i][j][r], lo);
        }
      }
  }
}

// ---------------------------------------------------------------------------
// weight repack: OIHW f32 -> [tap][O][Ip] f16 (zero pad c in [I, Ip))
// ---------------------------------------------------------------------------
__global__ __launch_bounds__(256) void pack_weights_t(const float* __restrict__ w,
                                                      _Float16* __restrict__ wkT,
                                                      int O, int I, int Ip)
{
  const int total = 9 * O * Ip;
  const int i = blockIdx.x * 256 + threadIdx.x;
  if (i >= total) return;
  const int r   = i / (O * Ip);
  const int rem = i - r * (O * Ip);
  const int o   = rem / Ip;
  const int c   = rem - o * Ip;
  const float v = (c < I) ? w[((size_t)o * I + c) * 9 + r] : 0.0f;
  wkT[i] = (_Float16)v;
}

// f32 NCHW -> f16 NHWC with channel padding (zeros for c >= Cin)
__global__ __launch_bounds__(256) void f32_to_nhwc_pad(const float* __restrict__ in,
                                                       _Float16* __restrict__ out,
                                                       int N, int Cin, int Cp, int H, int W)
{
  const size_t total = (size_t)N * H * W * Cp;
  const size_t i = (size_t)blockIdx.x * 256 + threadIdx.x;
  if (i >= total) return;
  const int c = (int)(i % Cp);
  size_t p = i / Cp;
  const int x = (int)(p % W); p /= W;
  const int y = (int)(p % H);
  const int n = (int)(p / H);
  float v = 0.0f;
  if (c < Cin) v = in[(((size_t)n * Cin + c) * H + y) * W + x];
  out[i] = (_Float16)v;
}

// NHWC 2x2 max pool
__global__ __launch_bounds__(256) void maxpool2x2_nhwc(const _Float16* __restrict__ in,
                                                       _Float16* __restrict__ out,
                                                       int N, int C, int H, int W, int Ho, int Wo)
{
  const size_t total = (size_t)N * Ho * Wo * C;
  const size_t i = (size_t)blockIdx.x * 256 + threadIdx.x;
  if (i >= total) return;
  const int c = (int)(i % C);
  size_t p = i / C;
  const int x = (int)(p % Wo); p /= Wo;
  const int y = (int)(p % Ho);
  const int n = (int)(p / Ho);
  const _Float16* s = in + ((size_t)((size_t)n * H + 2 * y) * W + 2 * x) * C + c;
  const size_t rs = (size_t)W * C;
  const float a = (float)s[0], b = (float)s[C], d = (float)s[rs], e = (float)s[rs + C];
  out[i] = (_Float16)fmaxf(fmaxf(a, b), fmaxf(d, e));
}

// fc1: act is NHWC (B,16,16,512); reference flattens NCHW -> k = c*256 + (y*16+x)
__global__ __launch_bounds__(256) void fc1_dot(const _Float16* __restrict__ act,
                                               const float* __restrict__ w,
                                               const float* __restrict__ b,
                                               float* __restrict__ out, int B, int Out)
{
  const int In = 512 * 16 * 16;
  const int wid  = blockIdx.x * 8 + (threadIdx.x >> 5);
  const int lane = threadIdx.x & 31;
  if (wid >= B * Out) return;
  const int bi = wid / Out, oi = wid - bi * Out;
  const _Float16* a = act + (size_t)bi * In;
  const float*   wr = w + (size_t)oi * In;
  float s = 0.0f;
  for (int k = lane; k < In; k += 32) {
    const int c = k >> 8;        // channel (256 = 16*16 spatial positions)
    const int p = k & 255;       // y*16+x
    s += (float)a[(size_t)p * 512 + c] * wr[k];
  }
#pragma unroll
  for (int off = 16; off; off >>= 1) s += __shfl_down(s, off, 32);
  if (lane == 0) out[wid] = s + b[oi];
}

__global__ __launch_bounds__(256) void fc_dot_f32(const float* __restrict__ act,
                                                  const float* __restrict__ w,
                                                  const float* __restrict__ b,
                                                  float* __restrict__ out, int B, int In, int Out)
{
  const int wid  = blockIdx.x * 8 + (threadIdx.x >> 5);
  const int lane = threadIdx.x & 31;
  if (wid >= B * Out) return;
  const int bi = wid / Out, oi = wid - bi * Out;
  const float* a  = act + (size_t)bi * In;
  const float* wr = w + (size_t)oi * In;
  float s = 0.0f;
  for (int k = lane; k < In; k += 32) s += a[k] * wr[k];
#pragma unroll
  for (int off = 16; off; off >>= 1) s += __shfl_down(s, off, 32);
  if (lane == 0) out[wid] = s + b[oi];
}

__global__ __launch_bounds__(256) void sigmoid_k(const float* __restrict__ h,
                                                 float* __restrict__ sg, int n)
{
  const int i = blockIdx.x * 256 + threadIdx.x;
  if (i < n) sg[i] = 1.0f / (1.0f + __expf(-h[i]));
}

// y = clip((x^a * 255*b + 255*c)/255, 0, 1); a/b/c indexed by the pixel's 30-group
__global__ __launch_bounds__(256) void final_transform(const float* __restrict__ inp,
                                                       const float* __restrict__ sg,
                                                       float* __restrict__ out, int B)
{
  const int PER = 3 * 291600;          // 3*540*540
  const size_t total = (size_t)B * PER;
  const size_t i = (size_t)blockIdx.x * 256 + threadIdx.x;
  if (i >= total) return;
  const int b  = (int)(i / PER);
  const int i2 = (int)(i - (size_t)b * PER);
  const int p  = i2 % 291600;
  const int g  = p / 9720;             // 291600 / 30
  const float a  = sg[b * 90 + g];
  const float bb = sg[b * 90 + 30 + g];
  const float cc = sg[b * 90 + 60 + g];
  const float x = inp[i];
  float y = (powf(x, a) * 255.0f * bb + 255.0f * cc) * (1.0f / 255.0f);
  out[i] = fminf(fmaxf(y, 0.0f), 1.0f);
}

// ---------------------------------------------------------------------------
extern "C" void kernel_launch(void* const* d_in, const int* in_sizes, int n_in,
                              void* d_out, int out_size, void* d_ws, size_t ws_size,
                              hipStream_t stream)
{
  (void)in_sizes; (void)n_in; (void)out_size; (void)ws_size;
  const int Bn = 2, S = 540;
  const float* inp = (const float*)d_in[0];

  // 13 backbone convs + rc1(13) + rc2(14); layer0 consumes the 32-ch padded input
  const int ci[15]  = {32,64,64,128,128,256,256,256,512,512,512,512,512, 128, 256};
  const int ciw[15] = {3, 64,64,128,128,256,256,256,512,512,512,512,512, 128, 256}; // true Cin in weights
  const int co[15]  = {64,64,128,128,256,256,256,512,512,512,512,512,512, 256, 512};
  const float* wsrc[15];
  const float* bsrc[13];
  for (int i = 0; i < 13; ++i) { wsrc[i] = (const float*)d_in[1 + 2 * i]; bsrc[i] = (const float*)d_in[2 + 2 * i]; }
  wsrc[13] = (const float*)d_in[27];   // rc1_w
  wsrc[14] = (const float*)d_in[28];   // rc2_w

  // workspace bump allocator
  char* ws = (char*)d_ws;
  size_t cur = 0;
  auto alloc = [&](size_t bytes) -> char* {
    char* p = ws + cur;
    cur += (bytes + 255) & ~(size_t)255;
    return p;
  };
  _Float16* wk[15];
  for (int i = 0; i < 15; ++i) wk[i] = (_Float16*)alloc((size_t)9 * co[i] * ci[i] * sizeof(_Float16));
  _Float16* A  = (_Float16*)alloc((size_t)37324800 * 2);   // max act: 2*540*540*64
  _Float16* Bb = (_Float16*)alloc((size_t)37324800 * 2);
  _Float16* C  = (_Float16*)alloc((size_t)18662400 * 2);   // padded inp16 (2*540*540*32) / out1..3
  _Float16* D  = (_Float16*)alloc((size_t)9331200 * 2);    // rc residuals, out4
  float* h1 = (float*)alloc(512 * sizeof(float));
  float* h2 = (float*)alloc(256 * sizeof(float));
  float* h3 = (float*)alloc(192 * sizeof(float));
  float* sg = (float*)alloc(192 * sizeof(float));

  // pack weights to [tap][Cout][Cin(padded)] f16
  for (int i = 0; i < 15; ++i) {
    const int total = 9 * co[i] * ci[i];
    pack_weights_t<<<(total + 255) / 256, 256, 0, stream>>>(wsrc[i], wk[i], co[i], ciw[i], ci[i]);
  }
  // input: f32 NCHW -> f16 NHWC padded to 32 channels
  { const size_t n = (size_t)Bn * S * S * 32;
    f32_to_nhwc_pad<<<(int)((n + 255) / 256), 256, 0, stream>>>(inp, C, Bn, 3, 32, S, S); }

  auto conv = [&](const _Float16* in, int L, const float* bias, const _Float16* resid,
                  _Float16* out, int H, int relu) {
    const int Npix = Bn * H * H;
    dim3 g((Npix + TM - 1) / TM, co[L] / TN);
    conv3x3_wmma<<<g, 256, 0, stream>>>(in, wk[L], bias, resid, out,
                                        Bn, ci[L], H, H, co[L], relu);
  };
  auto pool = [&](const _Float16* in, _Float16* out, int C_, int H) {
    const int Ho = H / 2;
    const size_t total = (size_t)Bn * Ho * Ho * C_;
    maxpool2x2_nhwc<<<(int)((total + 255) / 256), 256, 0, stream>>>(in, out, Bn, C_, H, H, Ho, Ho);
  };

  conv(C, 0, bsrc[0], nullptr, A, 540, 1);
  conv(A, 1, bsrc[1], nullptr, Bb, 540, 1);
  pool(Bb, A, 64, 540);                          // A: 270x270x64
  conv(A, 2, bsrc[2], nullptr, Bb, 270, 1);
  conv(Bb, 3, bsrc[3], nullptr, A, 270, 1);
  pool(A, C, 128, 270);                          // C: out1 (135x135x128)
  conv(C, 4, bsrc[4], nullptr, A, 135, 1);
  conv(A, 5, bsrc[5], nullptr, Bb, 135, 1);
  conv(C, 13, nullptr, nullptr, D, 135, 0);      // rc1(out1), no bias/relu
  conv(Bb, 6, bsrc[6], D, A, 135, 1);            // relu(s1 + rc1)
  pool(A, C, 256, 135);                          // C: out2 (67x67x256)
  conv(C, 7, bsrc[7], nullptr, A, 67, 1);
  conv(A, 8, bsrc[8], nullptr, Bb, 67, 1);
  conv(C, 14, nullptr, nullptr, D, 67, 0);       // rc2(out2)
  conv(Bb, 9, bsrc[9], D, A, 67, 1);             // relu(s2 + rc2)
  pool(A, C, 512, 67);                           // C: out3 (33x33x512)
  conv(C, 10, bsrc[10], nullptr, A, 33, 1);
  conv(A, 11, bsrc[11], nullptr, Bb, 33, 1);
  conv(Bb, 12, bsrc[12], C, A, 33, 1);           // relu(s3 + out3)
  pool(A, D, 512, 33);                           // D: out4 (16x16x512 NHWC)

  // FC head (no inter-layer activation per reference)
  fc1_dot<<<(Bn * 256 + 7) / 8, 256, 0, stream>>>(D, (const float*)d_in[29], (const float*)d_in[30],
                                                  h1, Bn, 256);
  fc_dot_f32<<<(Bn * 128 + 7) / 8, 256, 0, stream>>>(h1, (const float*)d_in[31], (const float*)d_in[32],
                                                     h2, Bn, 256, 128);
  fc_dot_f32<<<(Bn * 90 + 7) / 8, 256, 0, stream>>>(h2, (const float*)d_in[33], (const float*)d_in[34],
                                                    h3, Bn, 128, 90);
  sigmoid_k<<<1, 256, 0, stream>>>(h3, sg, Bn * 90);

  { const size_t total = (size_t)Bn * 3 * S * S;
    final_transform<<<(int)((total + 255) / 256), 256, 0, stream>>>(inp, sg, (float*)d_out, Bn); }
}
